// MambaBlock_65876208386194
// MI455X (gfx1250) — matrix operand order repaired and checked by reading, not compiled, not run on hardware
//
#include <hip/hip_runtime.h>

// ---------------------------------------------------------------------------
// Mamba block for MI455X (gfx1250): bf16 WMMA GEMMs + fp32 scan.
// ---------------------------------------------------------------------------

#define BATCH   2
#define LENGTH  2048
#define D_MODEL 1024
#define D_INNER 2048
#define DT_RANK 64
#define D_STATE 16
#define XDBL_W  (DT_RANK + 2 * D_STATE)   // 96
#define MTOT    (BATCH * LENGTH)          // 4096

typedef __attribute__((ext_vector_type(16))) __bf16 v16bf;
typedef __attribute__((ext_vector_type(8)))  __bf16 v8bf;
typedef __attribute__((ext_vector_type(8)))  float  v8f;

// ---- fp32 -> bf16, round-to-nearest-even ----------------------------------
__device__ __forceinline__ __bf16 f2bf(float f) {
    unsigned u = __builtin_bit_cast(unsigned, f);
    unsigned r = u + 0x7FFFu + ((u >> 16) & 1u);
    unsigned short h = (unsigned short)(r >> 16);
    return __builtin_bit_cast(__bf16, h);
}

__device__ __forceinline__ float softplusf(float x) {
    return x > 20.f ? x : log1pf(__expf(x));
}

// ---- elementwise fp32 -> bf16 copy ----------------------------------------
__global__ void cvt_bf16_kernel(const float* __restrict__ in,
                                __bf16* __restrict__ out, int n) {
    int i = blockIdx.x * blockDim.x + threadIdx.x;
    if (i < n) out[i] = f2bf(in[i]);
}

// ---- transpose + convert: in[K][N] f32 -> out[N][K] bf16 -------------------
__global__ void cvt_bf16_T_kernel(const float* __restrict__ in,
                                  __bf16* __restrict__ out, int K, int N) {
    int i = blockIdx.x * blockDim.x + threadIdx.x;
    if (i >= N * K) return;
    int n = i / K;
    int k = i % K;
    out[i] = f2bf(in[(size_t)k * N + n]);
}

// ---- extract dt_raw columns [0..64) from x_dbl [M,96] -> bf16 [M,64] -------
__global__ void cvt_dtraw_kernel(const float* __restrict__ xdbl,
                                 __bf16* __restrict__ out) {
    int i = blockIdx.x * blockDim.x + threadIdx.x;
    if (i >= MTOT * DT_RANK) return;
    int m = i / DT_RANK;
    int k = i % DT_RANK;
    out[i] = f2bf(xdbl[(size_t)m * XDBL_W + k]);
}

// ---- causal depthwise conv (ones(3)/3) + silu ------------------------------
__global__ void conv_silu_kernel(const float* __restrict__ xz,
                                 float* __restrict__ xc,
                                 __bf16* __restrict__ xcbf) {
    int i = blockIdx.x * blockDim.x + threadIdx.x;
    if (i >= BATCH * LENGTH * D_INNER) return;
    int d  = i % D_INNER;
    int t  = (i / D_INNER) % LENGTH;
    int bb = i / (D_INNER * LENGTH);
    size_t rb = (size_t)(bb * LENGTH + t) * (2 * D_INNER) + d;
    float s = xz[rb];
    if (t >= 1) s += xz[rb - (size_t)2 * D_INNER];
    if (t >= 2) s += xz[rb - (size_t)4 * D_INNER];
    float v = s * (1.f / 3.f);
    float sl = v / (1.f + __expf(-v));
    xc[i] = sl;
    xcbf[i] = f2bf(sl);
}

// ---------------------------------------------------------------------------
// bf16 WMMA GEMM: C[M,N] = A[M,K] * Bt[N,K]^T + bias, optional softplus.
// One wave computes a (MW*16) x (NW*16) tile: MW A-fragments are reused
// across NW B-fragments -> MW*NW WMMAs per 32-deep K-step.
// Fragment layouts per CDNA5 ISA 7.12.2 (wave32):
//   A 16x32 bf16: lane row = l&15, hi = l>>4; elems 0..7 = K[hi*8..+7],
//                 elems 8..15 = K[16+hi*8..+7]
//   B 32x16 bf16: lane col = l&15, hi = l>>4; same K order as A
//   C/D 16x16 f32: vgpr r, lane l -> M = r + 8*(l>>4), N = l&15
// ---------------------------------------------------------------------------
template <int MW, int NW>
__global__ void gemm_wmma_bf16_kernel(const __bf16* __restrict__ A,
                                      const __bf16* __restrict__ Bt,
                                      const float* __restrict__ bias,
                                      float* __restrict__ C,
                                      int M, int N, int K, int mode) {
    const int lane = threadIdx.x & 31;
    const int wid  = blockIdx.x * (blockDim.x >> 5) + (threadIdx.x >> 5);
    const int tilesM = M / (16 * MW);
    const int tilesN = N / (16 * NW);
    if (wid >= tilesM * tilesN) return;        // wave-uniform: EXEC stays all-1
    const int tm = wid % tilesM;
    const int tn = wid / tilesM;
    const int row0 = tm * 16 * MW;
    const int col0 = tn * 16 * NW;

    const int mrow = lane & 15;
    const int hi   = lane >> 4;

    const __bf16* aRow[MW];
    #pragma unroll
    for (int i = 0; i < MW; ++i)
        aRow[i] = A + (size_t)(row0 + i * 16 + mrow) * K;
    const __bf16* bRow[NW];
    #pragma unroll
    for (int j = 0; j < NW; ++j)
        bRow[j] = Bt + (size_t)(col0 + j * 16 + mrow) * K;

    v8f acc[MW][NW] = {};

    for (int k0 = 0; k0 < K; k0 += 32) {
        v16bf afr[MW];
        #pragma unroll
        for (int i = 0; i < MW; ++i) {
            v8bf alo = *(const v8bf*)(aRow[i] + k0 + hi * 8);
            v8bf ahi = *(const v8bf*)(aRow[i] + k0 + 16 + hi * 8);
            afr[i] = __builtin_shufflevector(alo, ahi,
                0, 1, 2, 3, 4, 5, 6, 7, 8, 9, 10, 11, 12, 13, 14, 15);
        }
        v16bf bfr[NW];
        #pragma unroll
        for (int j = 0; j < NW; ++j) {
            v8bf blo = *(const v8bf*)(bRow[j] + k0 + hi * 8);
            v8bf bhi = *(const v8bf*)(bRow[j] + k0 + 16 + hi * 8);
            bfr[j] = __builtin_shufflevector(blo, bhi,
                0, 1, 2, 3, 4, 5, 6, 7, 8, 9, 10, 11, 12, 13, 14, 15);
        }
        #pragma unroll
        for (int i = 0; i < MW; ++i)
            #pragma unroll
            for (int j = 0; j < NW; ++j)
                acc[i][j] = __builtin_amdgcn_wmma_f32_16x16x32_bf16(
                    false, afr[i], false, bfr[j], (short)0, acc[i][j],
                    false, false);
    }

    const int mbase = hi << 3;
    #pragma unroll
    for (int i = 0; i < MW; ++i) {
        #pragma unroll
        for (int j = 0; j < NW; ++j) {
            const int nn = col0 + j * 16 + (lane & 15);
            const float bv = bias[nn];
            #pragma unroll
            for (int r = 0; r < 8; ++r) {
                int mm = row0 + i * 16 + mbase + r;
                float v = acc[i][j][r] + bv;
                if (mode == 1) v = softplusf(v);
                C[(size_t)mm * N + nn] = v;
            }
        }
    }
}

// ---------------------------------------------------------------------------
// Selective scan: one thread per (batch, channel). h[16] + A-row in registers,
// B_t/C_t staged through LDS in 64-step chunks. Fuses +x_conv*D, silu(z) gate,
// and bf16 conversion for GEMM4.
// ---------------------------------------------------------------------------
__global__ void scan_kernel(const float* __restrict__ xdbl,   // [M,96]
                            const float* __restrict__ dtf,    // [M,D_INNER]
                            const float* __restrict__ xconv,  // [M,D_INNER]
                            const float* __restrict__ xz,     // [M,2*D_INNER]
                            const float* __restrict__ Alog,   // [D_INNER,16]
                            const float* __restrict__ Dv,     // [D_INNER]
                            __bf16* __restrict__ yin)         // [M,D_INNER]
{
    __shared__ float sBC[64 * 32];
    const int tid = threadIdx.x;
    const int chunksPerBatch = D_INNER / 256;
    const int b = blockIdx.x / chunksPerBatch;
    const int d = (blockIdx.x % chunksPerBatch) * 256 + tid;

    float Arow[D_STATE];
    float h[D_STATE];
    #pragma unroll
    for (int n = 0; n < D_STATE; ++n) {
        Arow[n] = -__expf(Alog[(size_t)d * D_STATE + n]);
        h[n] = 0.f;
    }
    const float Dd = Dv[d];

    for (int t0 = 0; t0 < LENGTH; t0 += 64) {
        __syncthreads();
        for (int j = tid; j < 64 * 32; j += 256) {
            int row = j >> 5, c = j & 31;
            sBC[j] = xdbl[(size_t)(b * LENGTH + t0 + row) * XDBL_W + DT_RANK + c];
        }
        __syncthreads();
        for (int tt = 0; tt < 64; ++tt) {
            const size_t base = (size_t)(b * LENGTH + t0 + tt);
            const float dtv = dtf[base * D_INNER + d];
            const float xv  = xconv[base * D_INNER + d];
            const float dx  = dtv * xv;
            const float* bc = &sBC[tt * 32];
            float y = 0.f;
            #pragma unroll
            for (int n = 0; n < D_STATE; ++n) {
                float dA = __expf(Arow[n] * dtv);
                h[n] = dA * h[n] + dx * bc[n];
                y += h[n] * bc[16 + n];
            }
            float yf = y + xv * Dd;
            float z = xz[base * (2 * D_INNER) + D_INNER + d];
            float g = z / (1.f + __expf(-z));
            yin[base * D_INNER + d] = f2bf(yf * g);
        }
    }
}

// ---------------------------------------------------------------------------
extern "C" void kernel_launch(void* const* d_in, const int* in_sizes, int n_in,
                              void* d_out, int out_size, void* d_ws, size_t ws_size,
                              hipStream_t stream) {
    (void)in_sizes; (void)n_in; (void)out_size; (void)ws_size;
    const float* x     = (const float*)d_in[0];
    const float* W_in  = (const float*)d_in[1];
    const float* b_in  = (const float*)d_in[2];
    const float* W_x   = (const float*)d_in[3];
    const float* b_x   = (const float*)d_in[4];
    const float* W_dt  = (const float*)d_in[5];
    const float* b_dt  = (const float*)d_in[6];
    const float* A_log = (const float*)d_in[7];
    const float* Dvec  = (const float*)d_in[8];
    const float* W_out = (const float*)d_in[9];
    const float* b_out = (const float*)d_in[10];
    float* out = (float*)d_out;

    // ---- workspace carve-up (256B aligned slices) ----
    char* w = (char*)d_ws;
    size_t off = 0;
    auto alloc = [&](size_t bytes) -> char* {
        char* p = w + off;
        off = (off + bytes + 255) & ~(size_t)255;
        return p;
    };
    __bf16* x_bf     = (__bf16*)alloc((size_t)MTOT * D_MODEL * 2);
    __bf16* WinT     = (__bf16*)alloc((size_t)(2 * D_INNER) * D_MODEL * 2);
    __bf16* WxT      = (__bf16*)alloc((size_t)XDBL_W * D_INNER * 2);
    __bf16* WdtT     = (__bf16*)alloc((size_t)D_INNER * DT_RANK * 2);
    __bf16* WoutT    = (__bf16*)alloc((size_t)D_MODEL * D_INNER * 2);
    float*  xzbuf    = (float*)alloc((size_t)MTOT * 2 * D_INNER * 4);
    float*  xconv    = (float*)alloc((size_t)MTOT * D_INNER * 4);
    __bf16* xconv_bf = (__bf16*)alloc((size_t)MTOT * D_INNER * 2);
    float*  xdbl     = (float*)alloc((size_t)MTOT * XDBL_W * 4);
    __bf16* dtraw_bf = (__bf16*)alloc((size_t)MTOT * DT_RANK * 2);
    float*  dtf      = (float*)alloc((size_t)MTOT * D_INNER * 4);
    __bf16* yin_bf   = (__bf16*)alloc((size_t)MTOT * D_INNER * 2);

    const int T = 256;
    auto blocks = [&](long long n) { return (unsigned)((n + T - 1) / T); };

    // 1) convert activations + weights to bf16 (weights transposed to [N,K])
    cvt_bf16_kernel<<<blocks((long long)MTOT * D_MODEL), T, 0, stream>>>(
        x, x_bf, MTOT * D_MODEL);
    cvt_bf16_T_kernel<<<blocks((long long)(2 * D_INNER) * D_MODEL), T, 0, stream>>>(
        W_in, WinT, D_MODEL, 2 * D_INNER);
    cvt_bf16_T_kernel<<<blocks((long long)XDBL_W * D_INNER), T, 0, stream>>>(
        W_x, WxT, D_INNER, XDBL_W);
    cvt_bf16_T_kernel<<<blocks((long long)D_INNER * DT_RANK), T, 0, stream>>>(
        W_dt, WdtT, DT_RANK, D_INNER);
    cvt_bf16_T_kernel<<<blocks((long long)D_MODEL * D_INNER), T, 0, stream>>>(
        W_out, WoutT, D_INNER, D_MODEL);

    // big-tile GEMM: 32x64 per wave (requires M%32==0, N%64==0)
    auto gemm_big = [&](const __bf16* A, const __bf16* Bt, const float* bias,
                        float* C, int M, int N, int K, int mode) {
        int waves = (M / 32) * (N / 64);
        int grid = (waves + 3) / 4;            // 4 waves (128 threads) per block
        gemm_wmma_bf16_kernel<2, 4><<<grid, 128, 0, stream>>>(
            A, Bt, bias, C, M, N, K, mode);
    };
    // small-tile GEMM: 16x32 per wave (for N=96)
    auto gemm_small = [&](const __bf16* A, const __bf16* Bt, const float* bias,
                          float* C, int M, int N, int K, int mode) {
        int waves = (M / 16) * (N / 32);
        int grid = (waves + 3) / 4;
        gemm_wmma_bf16_kernel<1, 2><<<grid, 128, 0, stream>>>(
            A, Bt, bias, C, M, N, K, mode);
    };

    // 2) xz = x @ W_in + b_in
    gemm_big(x_bf, WinT, b_in, xzbuf, MTOT, 2 * D_INNER, D_MODEL, 0);

    // 3) depthwise causal conv + silu
    conv_silu_kernel<<<blocks((long long)MTOT * D_INNER), T, 0, stream>>>(
        xzbuf, xconv, xconv_bf);

    // 4) x_dbl = x_conv @ W_x + b_x   (N = 96 = 3 * 32)
    gemm_small(xconv_bf, WxT, b_x, xdbl, MTOT, XDBL_W, D_INNER, 0);

    // 5) dt = softplus(dt_raw @ W_dt + b_dt)
    cvt_dtraw_kernel<<<blocks((long long)MTOT * DT_RANK), T, 0, stream>>>(
        xdbl, dtraw_bf);
    gemm_big(dtraw_bf, WdtT, b_dt, dtf, MTOT, D_INNER, DT_RANK, 1);

    // 6) selective scan + gate, fused epilogue -> bf16
    scan_kernel<<<BATCH * (D_INNER / 256), 256, 0, stream>>>(
        xdbl, dtf, xconv, xzbuf, A_log, Dvec, yin_bf);

    // 7) out = y_in @ W_out + b_out
    gemm_big(yin_bf, WoutT, b_out, out, MTOT, D_MODEL, D_INNER, 0);
}
